// LineAttention_71253507440758
// MI455X (gfx1250) — matrix operand order repaired and checked by reading, not compiled, not run on hardware
//
#include <hip/hip_runtime.h>

typedef unsigned short u16;
typedef unsigned int   u32;
typedef __bf16 bf16_t;
typedef bf16_t v16bf __attribute__((ext_vector_type(16)));
typedef float  v8f   __attribute__((ext_vector_type(8)));
typedef u32    u32x4 __attribute__((ext_vector_type(4)));

#define LINE    24
#define HEADS   8
#define DHEAD   64
#define DIM     256
#define INNER   512
#define NTOK    196608
#define LPB     2
#define ROWS    (LINE * LPB)   // 48
#define THREADS 256
#define NWAVE   8

// strides (in bf16/f32 elements)
#define XS_STR  264
#define QS_STR  520
#define VT_STR  56
#define SIM_STR 36
#define PB_STR  32

// LDS layout (bytes)
#define OFF_XS   0                 // xs (48*264*2=25344B), reused as os (48*520*2=49920B)
#define OFF_QS   49920             // 49920B
#define OFF_KS   99840             // 49920B
#define OFF_VT   149760            // 512*56*2 = 57344B
#define OFF_SIM  207104            // 8 * 32*36*4 = 36864B
#define OFF_PB   243968            // 8 * 32*32*2 = 16384B
#define LDS_BYTES 260352

__device__ __forceinline__ u16 f2bf(float f) {
  u32 u = __builtin_bit_cast(u32, f);
  u32 r = u + 0x7FFFu + ((u >> 16) & 1u);   // round-to-nearest-even
  return (u16)(r >> 16);
}

union FragBF { v16bf v; u32x4 q[2]; };

// A-matrix 16x32 bf16 fragment from a row-major bf16 buffer.
// Lane L: row = L&15; K chunks [k0 + (L>>4)*8 .. +8) and [k0+16+(L>>4)*8 .. +8).
__device__ __forceinline__ v16bf load_fragA(const u16* buf, int row0, int k0, int stride) {
  const int lane = threadIdx.x & 31;
  const u16* p = buf + (row0 + (lane & 15)) * stride + k0 + ((lane >> 4) * 8);
  FragBF f;
  f.q[0] = *(const u32x4*)(p);
  f.q[1] = *(const u32x4*)(p + 16);
  return f.v;
}

// B-matrix 32x16 bf16 fragment; bt is the *transposed* matrix [N][K] row-major.
// Lane L: column n0+(L&15); K chunks [k0 + (L>>4)*8 .. +8) and [k0+16+(L>>4)*8 .. +8).
__device__ __forceinline__ v16bf load_fragB(const u16* bt, int n0, int k0, int stride) {
  const int lane = threadIdx.x & 31;
  const u16* p = bt + (n0 + (lane & 15)) * stride + k0 + ((lane >> 4) * 8);
  FragBF f;
  f.q[0] = *(const u32x4*)(p);
  f.q[1] = *(const u32x4*)(p + 16);
  return f.v;
}

__device__ __forceinline__ v8f wmma_bf16(v16bf a, v16bf b, v8f c) {
  return __builtin_amdgcn_wmma_f32_16x16x32_bf16(false, a, false, b, (short)0, c, false, false);
}

// Convert f32 W[rows][cols] -> bf16 Wt[cols][rows]
__global__ void LineAttn_convT(const float* __restrict__ W, u16* __restrict__ Wt,
                               int rows, int cols) {
  int idx = blockIdx.x * blockDim.x + threadIdx.x;
  if (idx >= rows * cols) return;
  int r = idx / cols;
  int c = idx - r * cols;
  Wt[c * rows + r] = f2bf(W[idx]);
}

__global__ __launch_bounds__(THREADS)
void LineAttn_main(const float* __restrict__ x,
                   const u16* __restrict__ Wqt,    // [512][256]
                   const u16* __restrict__ Wkvt,   // [1024][256] (rows 0..511 = K, 512..1023 = V)
                   const u16* __restrict__ Woutt,  // [256][512]
                   const float* __restrict__ pos,  // [8][24][24]
                   float* __restrict__ out) {
  extern __shared__ char smem[];
  u16*   xs   = (u16*)(smem + OFF_XS);
  u16*   os   = (u16*)(smem + OFF_XS);   // reuse after stage 2
  u16*   qs   = (u16*)(smem + OFF_QS);
  u16*   ks   = (u16*)(smem + OFF_KS);
  u16*   vt   = (u16*)(smem + OFF_VT);   // V transposed [512][VT_STR tokens]
  float* simb = (float*)(smem + OFF_SIM);
  u16*   pb   = (u16*)(smem + OFF_PB);

  const int tid  = threadIdx.x;
  const int lane = tid & 31;
  const int wv   = tid >> 5;
  const long long rowbase = (long long)blockIdx.x * ROWS;

  // ---- stage 1: x tile -> bf16 LDS; zero V-transpose padding tokens 48..55 ----
  {
    const float* xg = x + rowbase * DIM;
    for (int e = tid; e < ROWS * DIM; e += THREADS) {
      int r = e >> 8;
      int c = e & 255;
      xs[r * XS_STR + c] = f2bf(xg[e]);
    }
    for (int e = tid; e < 512 * 8; e += THREADS) {
      int d = e >> 3;
      vt[d * VT_STR + 48 + (e & 7)] = 0;
    }
  }
  __syncthreads();

  // ---- stage 2: QKV projection [48,256] x [256,1536] ----
  for (int i = 0; i < 12; ++i) {
    const int nt = wv + i * 8;          // 0..95
    const int c0 = nt * 16;             // combined output column base
    const u16* brow = (c0 < 512) ? (Wqt + c0 * DIM) : (Wkvt + (c0 - 512) * DIM);
    v8f a0 = {}, a1 = {}, a2 = {};
    #pragma unroll
    for (int k = 0; k < 8; ++k) {
      v16bf B = load_fragB(brow, 0, k * 32, DIM);
      a0 = wmma_bf16(load_fragA(xs, 0,  k * 32, XS_STR), B, a0);
      a1 = wmma_bf16(load_fragA(xs, 16, k * 32, XS_STR), B, a1);
      a2 = wmma_bf16(load_fragA(xs, 32, k * 32, XS_STR), B, a2);
    }
    const int tb = 8 * (lane >> 4);
    if (c0 < 512) {                     // Q (scaled by DHEAD^-0.5 = 0.125)
      const int cn = c0 + (lane & 15);
      #pragma unroll
      for (int r = 0; r < 8; ++r) {
        qs[(tb + r)      * QS_STR + cn] = f2bf(a0[r] * 0.125f);
        qs[(tb + 16 + r) * QS_STR + cn] = f2bf(a1[r] * 0.125f);
        qs[(tb + 32 + r) * QS_STR + cn] = f2bf(a2[r] * 0.125f);
      }
    } else if (c0 < 1024) {             // K
      const int cn = c0 - 512 + (lane & 15);
      #pragma unroll
      for (int r = 0; r < 8; ++r) {
        ks[(tb + r)      * QS_STR + cn] = f2bf(a0[r]);
        ks[(tb + 16 + r) * QS_STR + cn] = f2bf(a1[r]);
        ks[(tb + 32 + r) * QS_STR + cn] = f2bf(a2[r]);
      }
    } else {                            // V, stored transposed [dim][token]
      const int d = c0 - 1024 + (lane & 15);
      #pragma unroll
      for (int r = 0; r < 8; ++r) {
        vt[d * VT_STR + tb + r]      = f2bf(a0[r]);
        vt[d * VT_STR + 16 + tb + r] = f2bf(a1[r]);
        vt[d * VT_STR + 32 + tb + r] = f2bf(a2[r]);
      }
    }
  }
  __syncthreads();

  // ---- stage 3: per-(line,head) attention, 2 pairs per wave ----
  for (int pp = 0; pp < 2; ++pp) {
    const int pr = wv + pp * 8;         // 0..15
    const int l  = pr >> 3;             // line in block
    const int h  = pr & 7;              // head
    const int r0 = l * LINE;            // token row base (0 or 24)
    float* sw = simb + wv * (32 * SIM_STR);
    u16*   pw = pb   + wv * (32 * PB_STR);

    // sim = Q * K^T  (M,N padded 24->32; K = 64)
    #pragma unroll
    for (int mt = 0; mt < 2; ++mt)
      #pragma unroll
      for (int ntt = 0; ntt < 2; ++ntt) {
        v8f acc = {};
        #pragma unroll
        for (int kk = 0; kk < 2; ++kk) {
          v16bf A = load_fragA(qs, r0 + mt * 16, h * DHEAD + kk * 32, QS_STR);
          v16bf B = load_fragB(ks, r0 + ntt * 16, h * DHEAD + kk * 32, QS_STR);
          acc = wmma_bf16(A, B, acc);
        }
        const int n  = ntt * 16 + (lane & 15);
        const int mb = mt * 16 + 8 * (lane >> 4);
        #pragma unroll
        for (int r = 0; r < 8; ++r)
          sw[(mb + r) * SIM_STR + n] = acc[r];
      }

    // zero prob buffer (so K padding 24..31 and M padding rows are exact zeros)
    {
      u32x4* pz = (u32x4*)pw;
      u32x4 z = 0;
      #pragma unroll
      for (int zi = 0; zi < 4; ++zi) pz[lane + zi * 32] = z;
    }

    // softmax: one row per lane (lanes 24..31 idle)
    if (lane < LINE) {
      const int i = lane;
      const float* pe = pos + (h * LINE + i) * LINE;
      float vals[LINE];
      float m = -1e30f;
      #pragma unroll
      for (int j = 0; j < LINE; ++j) {
        float s = sw[i * SIM_STR + j] + pe[j];
        vals[j] = s;
        m = fmaxf(m, s);
      }
      float sum = 0.f;
      #pragma unroll
      for (int j = 0; j < LINE; ++j) { float e = __expf(vals[j] - m); vals[j] = e; sum += e; }
      const float inv = 1.f / sum;
      #pragma unroll
      for (int j = 0; j < LINE; ++j) pw[i * PB_STR + j] = f2bf(vals[j] * inv);
    }

    // out = P * V   (K padded 24->32, zeros in P kill padded V tokens)
    #pragma unroll
    for (int mt = 0; mt < 2; ++mt)
      #pragma unroll
      for (int ntt = 0; ntt < 4; ++ntt) {
        v16bf A = load_fragA(pw, mt * 16, 0, PB_STR);
        v16bf B = load_fragB(vt, h * DHEAD + ntt * 16, r0, VT_STR);
        v8f acc = {};
        acc = wmma_bf16(A, B, acc);
        const int n  = h * DHEAD + ntt * 16 + (lane & 15);
        const int mb = mt * 16 + 8 * (lane >> 4);
        #pragma unroll
        for (int r = 0; r < 8; ++r) {
          const int local = mb + r;
          if (local < LINE) os[(r0 + local) * QS_STR + n] = f2bf(acc[r]);
        }
      }
  }
  __syncthreads();

  // ---- stage 4: out-projection [48,512] x [512,256] -> global f32 ----
  for (int t6 = 0; t6 < 6; ++t6) {
    const int id = wv + t6 * 8;         // 0..47
    const int mt = id >> 4;
    const int nt = id & 15;
    v8f acc = {};
    #pragma unroll
    for (int k = 0; k < 16; ++k) {
      v16bf A = load_fragA(os, mt * 16, k * 32, QS_STR);
      v16bf B = load_fragB(Woutt, nt * 16, k * 32, INNER);
      acc = wmma_bf16(A, B, acc);
    }
    const int col = nt * 16 + (lane & 15);
    const int mb  = mt * 16 + 8 * (lane >> 4);
    float* og = out + rowbase * DIM;
    #pragma unroll
    for (int r = 0; r < 8; ++r)
      og[(long long)(mb + r) * DIM + col] = acc[r];
  }
}

extern "C" void kernel_launch(void* const* d_in, const int* in_sizes, int n_in,
                              void* d_out, int out_size, void* d_ws, size_t ws_size,
                              hipStream_t stream) {
  const float* x    = (const float*)d_in[0];
  const float* Wq   = (const float*)d_in[1];
  const float* Wkv  = (const float*)d_in[2];
  const float* Wout = (const float*)d_in[3];
  const float* pos  = (const float*)d_in[4];
  float* out = (float*)d_out;

  u16* Wqt   = (u16*)d_ws;            // 512*256 bf16
  u16* Wkvt  = Wqt + 512 * 256;       // 1024*256 bf16
  u16* Woutt = Wkvt + 1024 * 256;     // 256*512 bf16

  LineAttn_convT<<<(512 * 256 + 255) / 256, 256, 0, stream>>>(Wq, Wqt, 256, 512);
  LineAttn_convT<<<(1024 * 256 + 255) / 256, 256, 0, stream>>>(Wkv, Wkvt, 256, 1024);
  LineAttn_convT<<<(512 * 256 + 255) / 256, 256, 0, stream>>>(Wout, Woutt, 512, 256);

  (void)hipFuncSetAttribute((const void*)LineAttn_main,
                            hipFuncAttributeMaxDynamicSharedMemorySize, LDS_BYTES);

  const int nblocks = (NTOK / LINE) / LPB;   // 4096
  LineAttn_main<<<nblocks, THREADS, LDS_BYTES, stream>>>(x, Wqt, Wkvt, Woutt, pos, out);
}
